// MYAttention_56169582297569
// MI455X (gfx1250) — matrix operand order repaired and checked by reading, not compiled, not run on hardware
//
#include <hip/hip_runtime.h>
#include <hip/hip_bf16.h>
#include <stdint.h>
#include <math.h>

// Problem sizes (fixed by the reference)
#define Bx 32
#define Nx 2048
#define Cx 128
#define Dx 1024
#define Ax 1024

typedef __bf16 bf16_t;
typedef __attribute__((ext_vector_type(16))) __bf16 v16bf;
typedef __attribute__((ext_vector_type(8)))  __bf16 v8bf;
typedef __attribute__((ext_vector_type(8)))  float  v8f;
typedef __attribute__((ext_vector_type(4)))  unsigned int u32x4;
typedef __attribute__((ext_vector_type(4)))  int i32x4;
typedef __attribute__((ext_vector_type(8)))  int i32x8;

// Hardware tanh (CDNA5 V_TANH_F32 trans op; co-executes with WMMA)
__device__ __forceinline__ float dev_tanh(float x) {
#if defined(__AMDGCN__) && __has_builtin(__builtin_amdgcn_tanhf)
  return __builtin_amdgcn_tanhf(x);
#elif defined(__AMDGCN__)
  float r;
  asm("v_tanh_f32 %0, %1" : "=v"(r) : "v"(x));
  return r;
#else
  return tanhf(x);
#endif
}

// ---------------------------------------------------------------------------
// Weight conversion f32 -> bf16 (wv_w is L2-resident afterwards: 2 MB)
// ---------------------------------------------------------------------------
__global__ __launch_bounds__(256)
void cvt_kernel(const float* __restrict__ in, bf16_t* __restrict__ out, int n) {
  int i = blockIdx.x * 256 + threadIdx.x;
  if (i < n) out[i] = (bf16_t)in[i];
}

// ---------------------------------------------------------------------------
// h[b,a] = dot(q[b,:], W[a,:]) + bias[a]   (tiny: 67 MFLOP)
// ---------------------------------------------------------------------------
__global__ __launch_bounds__(256)
void hproj_kernel(const float* __restrict__ q, const float* __restrict__ W,
                  const float* __restrict__ bias, float* __restrict__ out) {
  const int a = blockIdx.x * 256 + threadIdx.x;
  const int b = blockIdx.y;
  const float4* qr = (const float4*)(q + (size_t)b * Dx);
  const float4* wr = (const float4*)(W + (size_t)a * Dx);
  float acc = 0.f;
  for (int i = 0; i < Dx / 4; ++i) {
    float4 qa = qr[i], wv = wr[i];
    acc = fmaf(qa.x, wv.x, fmaf(qa.y, wv.y, fmaf(qa.z, wv.z, fmaf(qa.w, wv.w, acc))));
  }
  out[(size_t)b * Ax + a] = acc + bias[a];
}

// ---------------------------------------------------------------------------
// scores[b, n0+m] = sum_a tanh(h[b,a] + X[b,n0+m,:]·W[a,:] + vbias[a]) * wa[a]
// One block = 16 rows x A=1024, WMMA bf16 with f32 accumulation.
// Feats tile staged into LDS by the Tensor Data Mover (4-row chunks).
// ---------------------------------------------------------------------------
__global__ __launch_bounds__(256)
void score_kernel(const float* __restrict__ X, const bf16_t* __restrict__ Wbf,
                  const float* __restrict__ vbias, const float* __restrict__ hvec,
                  const float* __restrict__ wa, float* __restrict__ scores, int L) {
  __shared__ float  xstage[4 * Dx];      // 16 KB f32 staging (TDM target)
  __shared__ bf16_t xtile [16 * Dx];     // 32 KB bf16 A-tile
  __shared__ float  pscore[256][8];      // 8 KB partial scores (deterministic)

  const int tid  = threadIdx.x;
  const int wav  = tid >> 5;
  const int lane = tid & 31;
  const int half = lane >> 4;            // 0: lanes 0-15, 1: lanes 16-31
  const int nsub = lane & 15;
  const int b    = blockIdx.y;
  const int n0   = blockIdx.x * 16;
  const float* Xtile = X + ((size_t)b * L + n0) * Dx;

#if defined(__AMDGCN__) && __has_builtin(__builtin_amdgcn_tensor_load_to_lds)
  const unsigned lds_stage = (unsigned)(uintptr_t)&xstage[0];
  for (int ch = 0; ch < 4; ++ch) {
    if (wav == 0) {
      unsigned long long ga = (unsigned long long)(uintptr_t)(Xtile + ch * 4 * Dx);
      u32x4 g0 = {0u, 0u, 0u, 0u};
      i32x8 g1 = {0, 0, 0, 0, 0, 0, 0, 0};
      i32x4 gz4 = {0, 0, 0, 0};
      // D# group 0: count=1 | lds_addr | global_addr | type=2
      g0[0] = 1u;
      g0[1] = lds_stage;
      g0[2] = (unsigned)ga;
      g0[3] = (unsigned)((ga >> 32) & 0x1FFFFFFu) | (2u << 30);
      // D# group 1: data_size=4B, tensor 1024x4, tile 1024x4, row stride 1024
      g1[0] = (int)(2u << 16);               // data_size = 2 (4 bytes)
      g1[1] = (int)((unsigned)Dx << 16);     // tensor_dim0[15:0]  (bits 63:48)
      g1[2] = (int)(4u << 16);               // dim0 hi=0 | tensor_dim1=4
      g1[3] = (int)((unsigned)Dx << 16);     // dim1 hi=0 | tile_dim0=1024
      g1[4] = 4;                             // tile_dim1=4 | tile_dim2=0
      g1[5] = Dx;                            // tensor_dim0_stride[31:0]
#if __clang_major__ >= 23
      i32x8 gz8 = {0, 0, 0, 0, 0, 0, 0, 0};
      __builtin_amdgcn_tensor_load_to_lds(g0, g1, gz4, gz4, gz8, 0);
#else
      __builtin_amdgcn_tensor_load_to_lds(g0, g1, gz4, gz4, 0);
#endif
#if __has_builtin(__builtin_amdgcn_s_wait_tensorcnt)
      __builtin_amdgcn_s_wait_tensorcnt(0);
#else
      asm volatile("s_wait_tensorcnt 0" ::: "memory");
#endif
    }
    __syncthreads();
    for (int i = tid; i < 4 * Dx; i += 256)
      xtile[ch * 4 * Dx + i] = (bf16_t)xstage[i];
    __syncthreads();
  }
#else
  for (int i = tid; i < 16 * Dx; i += 256)
    xtile[i] = (bf16_t)Xtile[i];
  __syncthreads();
#endif

  float part[8];
#pragma unroll
  for (int v = 0; v < 8; ++v) part[v] = 0.f;

  const int koff = half * 8;               // A-frag K sub-offset per lane half
  const float* hrow = hvec + (size_t)b * Ax;

  for (int p = 0; p < 2; ++p) {
    const int abase = p * 512 + wav * 64;  // this wave's 64-column strip
    v8f acc[4] = {};
    for (int kk = 0; kk < Dx; kk += 32) {
      // A fragment: contiguous K order (lanes 0-15: K 0-15, 16-31: K 16-31)
      v8bf lo = *(const v8bf*)&xtile[nsub * Dx + kk + 2 * koff];
      v8bf hi = *(const v8bf*)&xtile[nsub * Dx + kk + 2 * koff + 8];
      v16bf afrag;
#pragma unroll
      for (int j = 0; j < 8; ++j) { afrag[j] = lo[j]; afrag[j + 8] = hi[j]; }
      const int kb = kk + half * 16;       // B 32x16: half-wave K split
#pragma unroll
      for (int t = 0; t < 4; ++t) {
        const int a = abase + t * 16 + nsub;
        v16bf bfrag = *(const v16bf*)&Wbf[(size_t)a * Dx + kb];
        acc[t] = __builtin_amdgcn_wmma_f32_16x16x32_bf16(
            false, afrag, false, bfrag, (short)0, acc[t], false, false);
      }
    }
    // epilogue: tanh(h + v + bias) * wa via hardware v_tanh_f32
#pragma unroll
    for (int t = 0; t < 4; ++t) {
      const int a = abase + t * 16 + nsub;
      const float hb = hrow[a] + vbias[a];
      const float w  = wa[a];
#pragma unroll
      for (int v = 0; v < 8; ++v) part[v] += dev_tanh(hb + acc[t][v]) * w;
    }
  }

#pragma unroll
  for (int v = 0; v < 8; ++v) pscore[tid][v] = part[v];
  __syncthreads();

  // Deterministic cross-wave reduction: thread m sums all lane partials for row m.
  if (tid < 16) {
    const int hb = (tid >> 3) << 4;        // rows 8-15 live in lanes 16-31
    float s = 0.f;
    for (int w = 0; w < 8; ++w)
      for (int n = 0; n < 16; ++n)
        s += pscore[w * 32 + hb + n][tid & 7];
    scores[(size_t)b * L + n0 + tid] = s;
  }
}

// ---------------------------------------------------------------------------
// In-place softmax over L per batch row (hw v_exp_f32 via __expf)
// ---------------------------------------------------------------------------
__global__ __launch_bounds__(256)
void softmax_kernel(float* __restrict__ s, int L) {
  __shared__ float red[256];
  const int b = blockIdx.x, tid = threadIdx.x;
  float m = -1e30f;
  for (int i = tid; i < L; i += 256) m = fmaxf(m, s[(size_t)b * L + i]);
  red[tid] = m;
  __syncthreads();
  for (int off = 128; off > 0; off >>= 1) {
    if (tid < off) red[tid] = fmaxf(red[tid], red[tid + off]);
    __syncthreads();
  }
  m = red[0];
  __syncthreads();
  float sum = 0.f;
  for (int i = tid; i < L; i += 256) sum += __expf(s[(size_t)b * L + i] - m);
  red[tid] = sum;
  __syncthreads();
  for (int off = 128; off > 0; off >>= 1) {
    if (tid < off) red[tid] = red[tid] + red[tid + off];
    __syncthreads();
  }
  const float inv = 1.f / red[0];
  for (int i = tid; i < L; i += 256)
    s[(size_t)b * L + i] = __expf(s[(size_t)b * L + i] - m) * inv;
}

// ---------------------------------------------------------------------------
// out[b,d] = (addin? addin[b,d] : 0) + sum_n alpha[b,n] * X[b,n,d]
// ---------------------------------------------------------------------------
__global__ __launch_bounds__(256)
void wsum_kernel(const float* __restrict__ X, const float* __restrict__ alpha,
                 const float* __restrict__ addin, float* __restrict__ out, int L) {
  __shared__ float sa[Nx];
  const int b = blockIdx.y;
  const int d = blockIdx.x * 256 + threadIdx.x;
  for (int i = threadIdx.x; i < L; i += 256) sa[i] = alpha[(size_t)b * L + i];
  __syncthreads();
  float acc = addin ? addin[(size_t)b * Dx + d] : 0.f;
  const float* Xb = X + (size_t)b * L * Dx + d;
  for (int n = 0; n < L; ++n) acc = fmaf(sa[n], Xb[(size_t)n * Dx], acc);
  out[(size_t)b * Dx + d] = acc;
}

// ---------------------------------------------------------------------------
extern "C" void kernel_launch(void* const* d_in, const int* in_sizes, int n_in,
                              void* d_out, int out_size, void* d_ws, size_t ws_size,
                              hipStream_t stream) {
  (void)in_sizes; (void)n_in; (void)out_size; (void)ws_size;
  const float* feats  = (const float*)d_in[0];
  const float* cfeats = (const float*)d_in[1];
  const float* key    = (const float*)d_in[2];
  const float* wh_w   = (const float*)d_in[3];
  const float* wh_b   = (const float*)d_in[4];
  const float* wv_w   = (const float*)d_in[5];
  const float* wv_b   = (const float*)d_in[6];
  const float* wa_w   = (const float*)d_in[7];
  const float* wh2_w  = (const float*)d_in[8];
  const float* wh2_b  = (const float*)d_in[9];
  const float* wv2_w  = (const float*)d_in[10];
  const float* wv2_b  = (const float*)d_in[11];
  const float* wa2_w  = (const float*)d_in[12];
  float* out = (float*)d_out;

  // Workspace layout (~4.7 MB total)
  bf16_t* wv_bf  = (bf16_t*)d_ws;
  bf16_t* wv2_bf = wv_bf + (size_t)Ax * Dx;
  float*  h1     = (float*)(wv2_bf + (size_t)Ax * Dx);
  float*  h2     = h1 + Bx * Ax;
  float*  sc1    = h2 + Bx * Ax;
  float*  sc2    = sc1 + Bx * Nx;
  float*  att1   = sc2 + Bx * Cx;

  const int nw = Ax * Dx;
  cvt_kernel<<<(nw + 255) / 256, 256, 0, stream>>>(wv_w,  wv_bf,  nw);
  cvt_kernel<<<(nw + 255) / 256, 256, 0, stream>>>(wv2_w, wv2_bf, nw);

  // Stage 1: feats (B,N,D), query = key
  hproj_kernel<<<dim3(Ax / 256, Bx), 256, 0, stream>>>(key, wh_w, wh_b, h1);
  score_kernel<<<dim3(Nx / 16, Bx), 256, 0, stream>>>(feats, wv_bf, wv_b, h1, wa_w, sc1, Nx);
  softmax_kernel<<<Bx, 256, 0, stream>>>(sc1, Nx);
  wsum_kernel<<<dim3(Dx / 256, Bx), 256, 0, stream>>>(feats, sc1, nullptr, att1, Nx);

  // Stage 2: cluster_feats (B,C,D), query = att1; out = att1 + att2
  hproj_kernel<<<dim3(Ax / 256, Bx), 256, 0, stream>>>(att1, wh2_w, wh2_b, h2);
  score_kernel<<<dim3(Cx / 16, Bx), 256, 0, stream>>>(cfeats, wv2_bf, wv2_b, h2, wa2_w, sc2, Cx);
  softmax_kernel<<<Bx, 256, 0, stream>>>(sc2, Cx);
  wsum_kernel<<<dim3(Dx / 256, Bx), 256, 0, stream>>>(cfeats, sc2, att1, out, Cx);
}
